// EnhancedMDSyn_11012296147007
// MI455X (gfx1250) — matrix operand order repaired and checked by reading, not compiled, not run on hardware
//
#include <hip/hip_runtime.h>
#include <math.h>

// ---------------- problem constants (match reference) ----------------
#define B_GR    256
#define V_NODES 128
#define NNODES  (B_GR * V_NODES)    // 32768
#define EPG_    512
#define NEDGE   (B_GR * EPG_)       // 131072
#define NETOT   (NEDGE + NNODES)    // edges + self loops
#define FIN     78
#define HID_    128
#define EPS_GN  1e-5f

typedef __attribute__((ext_vector_type(16))) _Float16 v16h;
typedef __attribute__((ext_vector_type(8)))  float    v8f;

#define ACT_NONE 0
#define ACT_RELU 1
#define ACT_SIGM 2

// =====================================================================
// WMMA GEMM: C[M,Nout] = act(A[M,K] @ W[K,Nout] + bias)
// Requires: M % 64 == 0, Nout % 64 == 0 (all layers except cl_w3).
// grid = (M/64, Nout/64), block = 128 (4 waves).
// Block tile = 64 rows x 64 cols; each wave owns 16 rows and 4 column
// sub-tiles (4x v8f accumulators). Double-buffered, software-pipelined
// B staging: load next W chunk into registers while WMMAs consume the
// current LDS buffer; one barrier per K-chunk. B tile is transposed
// f16 [col][k] so B fragments are 2x contiguous ds_load_b128 and
// staging stores are packed ds_store_b32. A fragments are unguarded
// float2 vector loads (all K even => rows 8B aligned, merged to b128).
// K tail handled branch-free with clamped addresses * {0,1}.
// =====================================================================
#define GEMM_BK 32
__global__ void gemm_wmma_kernel(const float* __restrict__ A,
                                 const float* __restrict__ W,
                                 const float* __restrict__ bias,
                                 float* __restrict__ C,
                                 int M, int K, int Nout, int act)
{
  __shared__ _Float16 lBt[2][64][GEMM_BK + 8];   // [buf][col][k], padded
  const int tid  = threadIdx.x;
  const int wave = tid >> 5;
  const int lane = tid & 31;
  const int row0 = blockIdx.x * 64 + wave * 16;
  const int colB = blockIdx.y * 64;
  const int half = (lane < 16) ? 0 : 1;
  const int ncol = lane & 15;
  const float* arow = A + (size_t)(row0 + ncol) * K;

  v8f acc[4] = {};
  float2 wv[8];     // staged W quads: [2*q] = row k0, [2*q+1] = row k0+1

  // staging geometry: item i -> k-pair k0 = (i>>5)*2, col-pair n0 = (i&31)*2
  auto ldW = [&](int kbase) {
#pragma unroll
    for (int q = 0; q < 4; ++q) {
      int i  = tid + q * 128;
      int k0 = (i >> 5) * 2;
      int n0 = (i & 31) * 2;
      const float* wp = &W[(size_t)(kbase + k0) * Nout + colB + n0];
      wv[2 * q]     = *(const float2*)wp;
      wv[2 * q + 1] = *(const float2*)(wp + Nout);
    }
  };
  auto stW = [&](int buf) {
#pragma unroll
    for (int q = 0; q < 4; ++q) {
      int i  = tid + q * 128;
      int k0 = (i >> 5) * 2;
      int n0 = (i & 31) * 2;
      union { _Float16 h[2]; unsigned u; } p0, p1;
      p0.h[0] = (_Float16)wv[2 * q].x;  p0.h[1] = (_Float16)wv[2 * q + 1].x;
      p1.h[0] = (_Float16)wv[2 * q].y;  p1.h[1] = (_Float16)wv[2 * q + 1].y;
      *(unsigned*)&lBt[buf][n0][k0]     = p0.u;
      *(unsigned*)&lBt[buf][n0 + 1][k0] = p1.u;
    }
  };
  auto mma4 = [&](int buf, const v16h& af) {
    int kc = half * 8;
#pragma unroll
    for (int nt = 0; nt < 4; ++nt) {
      v16h bf;
      const _Float16* bp = &lBt[buf][nt * 16 + ncol][kc];
#pragma unroll
      for (int i = 0; i < 16; ++i) bf[i] = bp[(i < 8) ? i : (i + 8)];
      acc[nt] = __builtin_amdgcn_wmma_f32_16x16x32_f16(false, af, false, bf,
                                                       (short)0, acc[nt],
                                                       false, false);
    }
  };

  const int Kfull = K & ~(GEMM_BK - 1);
  int kk = 0;
  int buf = 0;
  if (Kfull > 0) {
    ldW(0);
    for (kk = 0; kk < Kfull; kk += GEMM_BK) {
      stW(buf);
      if (kk + GEMM_BK < Kfull) ldW(kk + GEMM_BK);   // pipeline next W chunk
      __syncthreads();
      if (kk + GEMM_BK < K) __builtin_prefetch(arow + kk + GEMM_BK, 0, 0);
      // ---- A fragment: lanes0-15 K kk+0..7 / kk+16..23,
      //      lanes16-31 K kk+8..15 / kk+24..31 (row = lane&15) ----
      v16h af;
      {
        int kb = kk + half * 8;
        const float2* a0 = (const float2*)(arow + kb);
        const float2* a1 = (const float2*)(arow + kb + 16);
        float2 q0 = a0[0], q1 = a0[1], q2 = a0[2], q3 = a0[3];
        float2 r0 = a1[0], r1 = a1[1], r2 = a1[2], r3 = a1[3];
        af[0]  = (_Float16)q0.x; af[1]  = (_Float16)q0.y;
        af[2]  = (_Float16)q1.x; af[3]  = (_Float16)q1.y;
        af[4]  = (_Float16)q2.x; af[5]  = (_Float16)q2.y;
        af[6]  = (_Float16)q3.x; af[7]  = (_Float16)q3.y;
        af[8]  = (_Float16)r0.x; af[9]  = (_Float16)r0.y;
        af[10] = (_Float16)r1.x; af[11] = (_Float16)r1.y;
        af[12] = (_Float16)r2.x; af[13] = (_Float16)r2.y;
        af[14] = (_Float16)r3.x; af[15] = (_Float16)r3.y;
      }
      mma4(buf, af);
      buf ^= 1;
    }
  }

  // ---- K tail (< 32), branch-free via clamped loads ----
  if (kk < K) {
    __syncthreads();
#pragma unroll
    for (int q = 0; q < 4; ++q) {
      int i  = tid + q * 128;
      int k0 = (i >> 5) * 2;
      int n0 = (i & 31) * 2;
      int g0 = kk + k0, g1 = kk + k0 + 1;
      float s0 = (g0 < K) ? 1.0f : 0.0f;
      float s1 = (g1 < K) ? 1.0f : 0.0f;
      float2 a = *(const float2*)&W[(size_t)((g0 < K) ? g0 : K - 1) * Nout + colB + n0];
      float2 b = *(const float2*)&W[(size_t)((g1 < K) ? g1 : K - 1) * Nout + colB + n0];
      a.x *= s0; a.y *= s0; b.x *= s1; b.y *= s1;
      wv[2 * q] = a; wv[2 * q + 1] = b;
    }
    stW(buf);
    __syncthreads();
    v16h af;
#pragma unroll
    for (int i = 0; i < 8; ++i) {
      int g0 = kk + half * 8 + i;
      int g1 = g0 + 16;
      float v0 = arow[(g0 < K) ? g0 : (K - 1)] * ((g0 < K) ? 1.0f : 0.0f);
      float v1 = arow[(g1 < K) ? g1 : (K - 1)] * ((g1 < K) ? 1.0f : 0.0f);
      af[i]     = (_Float16)v0;
      af[i + 8] = (_Float16)v1;
    }
    mma4(buf, af);
  }

  // ---- writeback: VGPR r -> M = row0 + r + half*8; N = lane&15 ----
#pragma unroll
  for (int nt = 0; nt < 4; ++nt) {
    int n  = colB + nt * 16 + ncol;
    float bv = bias ? bias[n] : 0.0f;
#pragma unroll
    for (int r = 0; r < 8; ++r) {
      int m  = row0 + r + half * 8;
      float v = acc[nt][r] + bv;
      if (act == ACT_RELU)      v = fmaxf(v, 0.0f);
      else if (act == ACT_SIGM) v = 1.0f / (1.0f + __expf(-v));
      C[(size_t)m * Nout + n] = v;
    }
  }
}

// small fallback GEMM (one thread per output element) for Nout%64 != 0
__global__ void gemm_small_kernel(const float* __restrict__ A,
                                  const float* __restrict__ W,
                                  const float* __restrict__ bias,
                                  float* __restrict__ C,
                                  int M, int K, int Nout, int act)
{
  int i = blockIdx.x * blockDim.x + threadIdx.x;
  if (i >= M * Nout) return;
  int m = i / Nout, n = i % Nout;
  float acc = bias ? bias[n] : 0.0f;
  const float* ar = A + (size_t)m * K;
  for (int k = 0; k < K; ++k) acc += ar[k] * W[(size_t)k * Nout + n];
  if (act == ACT_RELU) acc = fmaxf(acc, 0.0f);
  C[i] = acc;
}

// =====================================================================
// elementwise / utility kernels
// =====================================================================
__global__ void fill_kernel(float* p, float v, long n) {
  long i = (long)blockIdx.x * blockDim.x + threadIdx.x;
  if (i < n) p[i] = v;
}
__global__ void copy_kernel(float* d, const float* s, long n) {
  long i = (long)blockIdx.x * blockDim.x + threadIdx.x;
  if (i < n) d[i] = s[i];
}
__global__ void mul_ip_kernel(float* y, const float* s, long n) {
  long i = (long)blockIdx.x * blockDim.x + threadIdx.x;
  if (i < n) y[i] *= s[i];
}
__global__ void bias_act_kernel(float* x, const float* __restrict__ bias,
                                long M, int C, int act) {
  long i = (long)blockIdx.x * blockDim.x + threadIdx.x;
  if (i >= M * C) return;
  int c = (int)(i % C);
  float v = x[i] + bias[c];
  if (act == ACT_RELU) v = fmaxf(v, 0.0f);
  x[i] = v;
}
__global__ void rsqrt_ip_kernel(float* p, long n) {
  long i = (long)blockIdx.x * blockDim.x + threadIdx.x;
  if (i < n) p[i] = rsqrtf(p[i]);
}

// =====================================================================
// graph scatter kernels (global float atomics)
// =====================================================================
__global__ void deg_scatter_kernel(const int* __restrict__ dst, float* deg, int ne) {
  int i = blockIdx.x * blockDim.x + threadIdx.x;
  if (i < ne) atomicAdd(&deg[dst[i]], 1.0f);
}
// out[dst] += h[src] * dinv[src]*dinv[dst]   (grid = NEDGE)
__global__ void gcn_edge_scatter_kernel(const float* __restrict__ h,
                                        const int* __restrict__ src,
                                        const int* __restrict__ dst,
                                        const float* __restrict__ dinv,
                                        float* out, int C) {
  int e = blockIdx.x;
  int s = src[e], d = dst[e];
  float nrm = dinv[s] * dinv[d];
  const float* hs = h + (size_t)s * C;
  float* od = out + (size_t)d * C;
  for (int c = threadIdx.x; c < C; c += blockDim.x)
    atomicAdd(&od[c], hs[c] * nrm);
}
__global__ void gcn_self_kernel(const float* __restrict__ h,
                                const float* __restrict__ dinv,
                                float* out, long n, int C) {
  long i = (long)blockIdx.x * blockDim.x + threadIdx.x;
  if (i >= n * C) return;
  long r = i / C;
  float di = dinv[r];
  out[i] += h[i] * di * di;
}
// out[dst] += h[src]   (grid = NEDGE)
__global__ void edge_scatter_add_kernel(const float* __restrict__ h,
                                        const int* __restrict__ src,
                                        const int* __restrict__ dst,
                                        float* out, int C) {
  int e = blockIdx.x;
  int s = src[e], d = dst[e];
  const float* hs = h + (size_t)s * C;
  float* od = out + (size_t)d * C;
  for (int c = threadIdx.x; c < C; c += blockDim.x)
    atomicAdd(&od[c], hs[c]);
}

// =====================================================================
// GATv2 (edges 0..NE-1 from arrays; edges NE..NE+N-1 are self loops)
// =====================================================================
__device__ inline void atomicMaxF(float* addr, float val) {
  if (val >= 0.0f) atomicMax((int*)addr, __float_as_int(val));
  else             atomicMin((unsigned int*)addr, __float_as_uint(val));
}
__global__ void gat_logits_kernel(const float* __restrict__ xl,
                                  const float* __restrict__ xr,
                                  const float* __restrict__ att,
                                  const int* __restrict__ src,
                                  const int* __restrict__ dst,
                                  float* logits, float* maxb,
                                  int NE, int Nn, int H, int Ch) {
  int idx = blockIdx.x * blockDim.x + threadIdx.x;
  int tot = (NE + Nn) * H;
  if (idx >= tot) return;
  int e = idx / H, h = idx % H;
  int s = (e < NE) ? src[e] : (e - NE);
  int d = (e < NE) ? dst[e] : (e - NE);
  const float* a = att + h * Ch;
  const float* l = xl + (size_t)s * H * Ch + h * Ch;
  const float* r = xr + (size_t)d * H * Ch + h * Ch;
  float acc = 0.0f;
  for (int c = 0; c < Ch; ++c) {
    float v = l[c] + r[c];
    v = (v > 0.0f) ? v : 0.2f * v;        // leaky_relu(0.2)
    acc += a[c] * v;
  }
  logits[idx] = acc;
  atomicMaxF(&maxb[d * H + h], acc);
}
__global__ void gat_exp_kernel(float* logits, const float* __restrict__ maxb,
                               float* den, const int* __restrict__ dst,
                               int NE, int Nn, int H) {
  int idx = blockIdx.x * blockDim.x + threadIdx.x;
  int tot = (NE + Nn) * H;
  if (idx >= tot) return;
  int e = idx / H, h = idx % H;
  int d = (e < NE) ? dst[e] : (e - NE);
  float p = __expf(logits[idx] - maxb[d * H + h]);
  logits[idx] = p;
  atomicAdd(&den[d * H + h], p);
}
// out[d] += (p/den[d]) * xl[s]   (grid = NETOT, block = H*Ch)
__global__ void gat_scatter_kernel(const float* __restrict__ xl,
                                   const float* __restrict__ logits,
                                   const float* __restrict__ den,
                                   const int* __restrict__ src,
                                   const int* __restrict__ dst,
                                   float* out, int NE, int Nn, int H, int Ch) {
  int e = blockIdx.x;
  int t = threadIdx.x;          // 0..H*Ch-1
  int h = t / Ch;
  int s = (e < NE) ? src[e] : (e - NE);
  int d = (e < NE) ? dst[e] : (e - NE);
  float alpha = logits[e * H + h] / den[d * H + h];
  atomicAdd(&out[(size_t)d * H * Ch + t], alpha * xl[(size_t)s * H * Ch + t]);
}

// =====================================================================
// GraphNorm (C=512, per-graph stats, count = V_NODES)
// =====================================================================
__global__ void gn_sum_kernel(const float* __restrict__ x, float* acc, int C) {
  int nID = blockIdx.x;
  int b = nID / V_NODES;
  const float* xr = x + (size_t)nID * C;
  float* m = acc + (size_t)b * C;
  for (int c = threadIdx.x; c < C; c += blockDim.x) atomicAdd(&m[c], xr[c]);
}
__global__ void gn_center_kernel(const float* __restrict__ x,
                                 const float* __restrict__ gmean,
                                 const float* __restrict__ ms,
                                 float* y, int C) {
  long i = (long)blockIdx.x * blockDim.x + threadIdx.x;
  if (i >= (long)NNODES * C) return;
  long r = i / C; int c = (int)(i % C);
  long b = r / V_NODES;
  y[i] = x[i] - (gmean[b * C + c] * (1.0f / V_NODES)) * ms[c];
}
__global__ void gn_sumsq_kernel(const float* __restrict__ y, float* acc, int C) {
  int nID = blockIdx.x;
  int b = nID / V_NODES;
  const float* yr = y + (size_t)nID * C;
  float* m = acc + (size_t)b * C;
  for (int c = threadIdx.x; c < C; c += blockDim.x) atomicAdd(&m[c], yr[c] * yr[c]);
}
__global__ void gn_norm_kernel(float* y, const float* __restrict__ gvar,
                               const float* __restrict__ w,
                               const float* __restrict__ bb, int C) {
  long i = (long)blockIdx.x * blockDim.x + threadIdx.x;
  if (i >= (long)NNODES * C) return;
  long r = i / C; int c = (int)(i % C);
  long b = r / V_NODES;
  y[i] = y[i] * rsqrtf(gvar[b * C + c] * (1.0f / V_NODES) + EPS_GN) * w[c] + bb[c];
}

// =====================================================================
// BatchNorm over M=256 rows (grid = C, block = 256 = M), in place
// =====================================================================
__global__ void bn_kernel(float* x, const float* __restrict__ g,
                          const float* __restrict__ b, int M, int C, int act) {
  __shared__ float sb[256];
  int c = blockIdx.x, t = threadIdx.x;
  float v = x[(size_t)t * C + c];
  sb[t] = v;
  __syncthreads();
  for (int s = 128; s > 0; s >>= 1) { if (t < s) sb[t] += sb[t + s]; __syncthreads(); }
  float mean = sb[0] / M;
  __syncthreads();
  float dv = v - mean;
  sb[t] = dv * dv;
  __syncthreads();
  for (int s = 128; s > 0; s >>= 1) { if (t < s) sb[t] += sb[t + s]; __syncthreads(); }
  float var = sb[0] / M;
  float y = g[c] * dv * rsqrtf(var + EPS_GN) + b[c];
  if (act == ACT_RELU) y = fmaxf(y, 0.0f);
  x[(size_t)t * C + c] = y;
}

// =====================================================================
// concat / pooling / final
// =====================================================================
__global__ void concat3_kernel(const float* g, const float* h, const float* a,
                               float* multi) {
  long i = (long)blockIdx.x * blockDim.x + threadIdx.x;
  if (i >= (long)NNODES * 384) return;
  long r = i / 384; int c = (int)(i % 384);
  float v;
  if (c < 128)      v = g[r * 128 + c];
  else if (c < 256) v = h[r * 128 + c - 128];
  else              v = a[r * 128 + c - 256];
  multi[i] = v;
}
__global__ void concat5_kernel(const float* pe1, const float* pe2, const float* cell,
                               const float* d1, const float* d2, float* comb) {
  long i = (long)blockIdx.x * blockDim.x + threadIdx.x;
  if (i >= (long)B_GR * 1024) return;
  long r = i / 1024; int c = (int)(i % 1024);
  float v;
  if (c < 256)      v = pe1[r * 256 + c];
  else if (c < 512) v = pe2[r * 256 + c - 256];
  else if (c < 768) v = cell[r * 256 + c - 512];
  else if (c < 896) v = d1[r * 128 + c - 768];
  else              v = d2[r * 128 + c - 896];
  comb[i] = v;
}
__global__ void mean_pool_kernel(const float* __restrict__ fused, float* out) {
  int b = blockIdx.x, c = threadIdx.x;   // block = 128
  float s = 0.0f;
  for (int v = 0; v < V_NODES; ++v)
    s += fused[((size_t)b * V_NODES + v) * HID_ + c];
  out[(size_t)b * HID_ + c] = s * (1.0f / V_NODES);
}
__global__ void final_kernel(const float* __restrict__ h, const float* __restrict__ w,
                             const float* __restrict__ b, float* out) {
  int i = blockIdx.x * blockDim.x + threadIdx.x;
  if (i >= B_GR) return;
  const float* hr = h + (size_t)i * 32;
  for (int j = 0; j < 2; ++j) {
    float acc = b[j];
    for (int k = 0; k < 32; ++k) acc += hr[k] * w[k * 2 + j];
    out[i * 2 + j] = acc;
  }
  out[B_GR * 2 + i] = 1.0f;   // dummy_weight
}

// =====================================================================
// host orchestration
// =====================================================================
#define PRM(i) ((const float*)d_in[10 + (i)])

extern "C" void kernel_launch(void* const* d_in, const int* in_sizes, int n_in,
                              void* d_out, int out_size, void* d_ws, size_t ws_size,
                              hipStream_t stream)
{
  (void)in_sizes; (void)n_in; (void)out_size; (void)ws_size;
  const float* x1   = (const float*)d_in[0];
  const int*   ei1  = (const int*)d_in[1];
  const float* sm1  = (const float*)d_in[3];
  const float* ccle = (const float*)d_in[4];
  const float* x2   = (const float*)d_in[5];
  const int*   ei2  = (const int*)d_in[6];
  const float* sm2  = (const float*)d_in[8];
  float* out = (float*)d_out;

  const long N = NNODES;
  float* wsp = (float*)d_ws;
  size_t off = 0;
  auto alloc = [&](size_t n) { float* p = wsp + off; off += n; return p; };

  float* T0    = alloc((size_t)N * 512);
  float* T1    = alloc((size_t)N * 512);
  float* T2    = alloc((size_t)N * 512);
  float* G128a = alloc((size_t)N * 128);
  float* G128b = alloc((size_t)N * 128);
  float* H128a = alloc((size_t)N * 128);
  float* HOUT  = alloc((size_t)N * 128);
  float* XL2   = alloc((size_t)N * 128);
  float* XR2   = alloc((size_t)N * 128);
  float* AOUT  = alloc((size_t)N * 128);
  float* MULTI = alloc((size_t)N * 384);
  float* T192  = alloc((size_t)N * 192);
  float* T384  = alloc((size_t)N * 384);
  float* FUSED = alloc((size_t)N * 128);
  float* AGG78 = alloc((size_t)N * FIN);
  float* DEG   = alloc(N);
  float* LOGI  = alloc((size_t)NETOT * 2);
  float* MAXB  = alloc((size_t)N * 2);
  float* DEN   = alloc((size_t)N * 2);
  float* GMEAN = alloc((size_t)B_GR * 512);
  float* GVAR  = alloc((size_t)B_GR * 512);
  float* D1    = alloc((size_t)B_GR * 128);
  float* D2    = alloc((size_t)B_GR * 128);
  float* E512  = alloc((size_t)B_GR * 512);
  float* PE1   = alloc((size_t)B_GR * 256);
  float* PE2   = alloc((size_t)B_GR * 256);
  float* CELL  = alloc((size_t)B_GR * 256);
  float* COMB  = alloc((size_t)B_GR * 1024);
  float* C512  = alloc((size_t)B_GR * 512);
  float* C128  = alloc((size_t)B_GR * 128);
  float* C32   = alloc((size_t)B_GR * 32);

  auto gemm = [&](const float* A, const float* W, const float* bias, float* C,
                  int M, int K, int Nout, int act) {
    if ((Nout & 63) == 0) {
      dim3 g(M / 64, Nout / 64);
      gemm_wmma_kernel<<<g, 128, 0, stream>>>(A, W, bias, C, M, K, Nout, act);
    } else {
      gemm_small_kernel<<<(M * Nout + 255) / 256, 256, 0, stream>>>(A, W, bias, C,
                                                                    M, K, Nout, act);
    }
  };
  auto fill = [&](float* p, float v, long n) {
    fill_kernel<<<(unsigned)((n + 255) / 256), 256, 0, stream>>>(p, v, n);
  };
  auto copyb = [&](float* d, const float* s, long n) {
    copy_kernel<<<(unsigned)((n + 255) / 256), 256, 0, stream>>>(d, s, n);
  };
  auto biasact = [&](float* x, const float* b, long M, int C, int act) {
    bias_act_kernel<<<(unsigned)((M * C + 255) / 256), 256, 0, stream>>>(x, b, M, C, act);
  };

  auto gatv2 = [&](const float* xl, const float* xr, const float* att,
                   const int* src, const int* dst, float* outb, int H, int Ch) {
    int tot = NETOT * H;
    fill(MAXB, -1e30f, (long)N * H);
    gat_logits_kernel<<<(tot + 255) / 256, 256, 0, stream>>>(xl, xr, att, src, dst,
                                                             LOGI, MAXB, NEDGE, N, H, Ch);
    fill(DEN, 0.0f, (long)N * H);
    gat_exp_kernel<<<(tot + 255) / 256, 256, 0, stream>>>(LOGI, MAXB, DEN, dst,
                                                          NEDGE, N, H);
    fill(outb, 0.0f, (long)N * H * Ch);
    gat_scatter_kernel<<<NETOT, H * Ch, 0, stream>>>(xl, LOGI, DEN, src, dst, outb,
                                                     NEDGE, N, H, Ch);
  };

  auto drug_branch = [&](const float* x, const int* ei, float* dPool) {
    const int* src = ei;
    const int* dst = ei + NEDGE;
    // ---- GCN branch ----
    gemm(x, PRM(0), nullptr, T0, N, FIN, 512, ACT_NONE);                  // x@gcn1_w
    fill(DEG, 1.0f, N);                                                   // self loop
    deg_scatter_kernel<<<(NEDGE + 255) / 256, 256, 0, stream>>>(dst, DEG, NEDGE);
    rsqrt_ip_kernel<<<(N + 255) / 256, 256, 0, stream>>>(DEG, N);         // dinv
    fill(T1, 0.0f, (long)N * 512);
    gcn_edge_scatter_kernel<<<NEDGE, 128, 0, stream>>>(T0, src, dst, DEG, T1, 512);
    gcn_self_kernel<<<(unsigned)(((long)N * 512 + 255) / 256), 256, 0, stream>>>(T0, DEG, T1, N, 512);
    biasact(T1, PRM(1), N, 512, ACT_RELU);
    gemm(T1, PRM(2), nullptr, G128a, N, 512, 128, ACT_NONE);              // @gcn2_w
    fill(G128b, 0.0f, (long)N * 128);
    gcn_edge_scatter_kernel<<<NEDGE, 128, 0, stream>>>(G128a, src, dst, DEG, G128b, 128);
    gcn_self_kernel<<<(unsigned)(((long)N * 128 + 255) / 256), 256, 0, stream>>>(G128a, DEG, G128b, N, 128);
    biasact(G128b, PRM(3), N, 128, ACT_NONE);
    // ---- GIN branch ----
    copyb(AGG78, x, (long)N * FIN);
    edge_scatter_add_kernel<<<NEDGE, 128, 0, stream>>>(x, src, dst, AGG78, FIN);
    gemm(AGG78, PRM(4), PRM(5), T0, N, FIN, 512, ACT_RELU);
    gemm(T0, PRM(6), PRM(7), T1, N, 512, 512, ACT_RELU);                  // relu before gnorm
    fill(GMEAN, 0.0f, (long)B_GR * 512);
    gn_sum_kernel<<<N, 256, 0, stream>>>(T1, GMEAN, 512);
    gn_center_kernel<<<(unsigned)(((long)N * 512 + 255) / 256), 256, 0, stream>>>(T1, GMEAN, PRM(14), T2, 512);
    fill(GVAR, 0.0f, (long)B_GR * 512);
    gn_sumsq_kernel<<<N, 256, 0, stream>>>(T2, GVAR, 512);
    gn_norm_kernel<<<(unsigned)(((long)N * 512 + 255) / 256), 256, 0, stream>>>(T2, GVAR, PRM(12), PRM(13), 512);
    copyb(T0, T2, (long)N * 512);
    edge_scatter_add_kernel<<<NEDGE, 128, 0, stream>>>(T2, src, dst, T0, 512);
    gemm(T0, PRM(8), PRM(9), H128a, N, 512, 128, ACT_RELU);
    gemm(H128a, PRM(10), PRM(11), HOUT, N, 128, 128, ACT_NONE);
    // ---- GATv2 branch ----
    gemm(x, PRM(15), PRM(16), T1, N, FIN, 512, ACT_NONE);                 // xl
    gemm(x, PRM(17), PRM(18), T2, N, FIN, 512, ACT_NONE);                 // xr
    gatv2(T1, T2, PRM(19), src, dst, T0, 2, 256);
    biasact(T0, PRM(20), N, 512, ACT_RELU);
    gemm(T0, PRM(21), PRM(22), XL2, N, 512, 128, ACT_NONE);
    gemm(T0, PRM(23), PRM(24), XR2, N, 512, 128, ACT_NONE);
    gatv2(XL2, XR2, PRM(25), src, dst, AOUT, 1, 128);
    biasact(AOUT, PRM(26), N, 128, ACT_NONE);
    // ---- fusion ----
    concat3_kernel<<<(unsigned)(((long)N * 384 + 255) / 256), 256, 0, stream>>>(G128b, HOUT, AOUT, MULTI);
    gemm(MULTI, PRM(27), PRM(28), T192, N, 384, 192, ACT_RELU);
    gemm(T192, PRM(29), PRM(30), T384, N, 192, 384, ACT_SIGM);
    mul_ip_kernel<<<(unsigned)(((long)N * 384 + 255) / 256), 256, 0, stream>>>(MULTI, T384, (long)N * 384);
    gemm(MULTI, PRM(31), PRM(32), FUSED, N, 384, 128, ACT_NONE);
    mean_pool_kernel<<<B_GR, 128, 0, stream>>>(FUSED, dPool);
  };

  // ---- encoders (drug embedding processor, shared weights; cell line) ----
  gemm(sm1, PRM(33), PRM(34), E512, B_GR, 768, 512, ACT_NONE);
  bn_kernel<<<512, 256, 0, stream>>>(E512, PRM(35), PRM(36), B_GR, 512, ACT_RELU);
  gemm(E512, PRM(37), PRM(38), PE1, B_GR, 512, 256, ACT_NONE);
  bn_kernel<<<256, 256, 0, stream>>>(PE1, PRM(39), PRM(40), B_GR, 256, ACT_RELU);

  gemm(sm2, PRM(33), PRM(34), E512, B_GR, 768, 512, ACT_NONE);
  bn_kernel<<<512, 256, 0, stream>>>(E512, PRM(35), PRM(36), B_GR, 512, ACT_RELU);
  gemm(E512, PRM(37), PRM(38), PE2, B_GR, 512, 256, ACT_NONE);
  bn_kernel<<<256, 256, 0, stream>>>(PE2, PRM(39), PRM(40), B_GR, 256, ACT_RELU);

  gemm(ccle, PRM(41), PRM(42), E512, B_GR, 954, 512, ACT_NONE);
  bn_kernel<<<512, 256, 0, stream>>>(E512, PRM(43), PRM(44), B_GR, 512, ACT_RELU);
  gemm(E512, PRM(45), PRM(46), CELL, B_GR, 512, 256, ACT_NONE);
  bn_kernel<<<256, 256, 0, stream>>>(CELL, PRM(47), PRM(48), B_GR, 256, ACT_RELU);

  // ---- drug graph branches ----
  drug_branch(x1, ei1, D1);
  drug_branch(x2, ei2, D2);

  // ---- classifier ----
  concat5_kernel<<<(unsigned)(((long)B_GR * 1024 + 255) / 256), 256, 0, stream>>>(PE1, PE2, CELL, D1, D2, COMB);
  gemm(COMB, PRM(49), PRM(50), C512, B_GR, 1024, 512, ACT_NONE);
  bn_kernel<<<512, 256, 0, stream>>>(C512, PRM(51), PRM(52), B_GR, 512, ACT_RELU);
  gemm(C512, PRM(53), PRM(54), C128, B_GR, 512, 128, ACT_NONE);
  bn_kernel<<<128, 256, 0, stream>>>(C128, PRM(55), PRM(56), B_GR, 128, ACT_RELU);
  gemm(C128, PRM(57), PRM(58), C32, B_GR, 128, 32, ACT_RELU);   // small path
  final_kernel<<<1, 256, 0, stream>>>(C32, PRM(59), PRM(60), out);
}
